// WanSelfAttention_16930761081101
// MI455X (gfx1250) — matrix operand order repaired and checked by reading, not compiled, not run on hardware
//
#include <hip/hip_runtime.h>
#include <hip/hip_bf16.h>

#define DIM    3072
#define NHEADS 24
#define HDIM   128
#define SEQ    2304
#define C2TAB  64     // freqs row length
#define CF_P   22     // frame-axis pairs
#define CH_P   21     // H-axis pairs
#define EPSF   1e-6f
#define GEMM_BK 128   // K-chunk staged per TDM transfer

typedef __attribute__((ext_vector_type(16))) __bf16 v16bf;
typedef __attribute__((ext_vector_type(8)))  float  v8f;
typedef __attribute__((ext_vector_type(4)))  unsigned int u32x4;
typedef __attribute__((ext_vector_type(8)))  int    i32x8;
typedef __attribute__((ext_vector_type(4)))  int    i32x4;

union FragBF { v16bf v; u32x4 u4[2]; unsigned short h[16]; };

__device__ __forceinline__ unsigned short f2bf(float f) {
  unsigned int u = __float_as_uint(f);
  unsigned int r = (u + 0x7FFFu + ((u >> 16) & 1u)) >> 16;   // round-to-nearest-even
  return (unsigned short)r;
}

// Opaque LDS fragment read: 32 contiguous bytes at byte-offset `off`.
// Hidden from the optimizer (TDM writes are invisible to alias analysis, so
// ordinary loads of the staged tile get folded to undef). The embedded
// s_wait_dscnt makes the results architecturally valid when the asm retires.
__device__ __forceinline__ void lds_read_frag32(unsigned off, u32x4& lo, u32x4& hi) {
  asm volatile("ds_load_b128 %0, %2\n\t"
               "ds_load_b128 %1, %2 offset:16\n\t"
               "s_wait_dscnt 0"
               : "=&v"(lo), "=&v"(hi)
               : "v"(off)
               : "memory");
}

// --------------------------------------------------------------- TDM tile load
// 2D tile (tile_k x tile_n) of bf16, row stride = row_stride elems, -> LDS.
// D# layout per CDNA5 ISA 8.3/8.4: group0 = {flags, lds_addr, gaddr_lo, gaddr_hi|type},
// group1 = {mask/data_size, dims/tiles/strides}. data_size=1 (2 bytes).
__device__ __forceinline__ void tdm_load_tile_bf16(
    const unsigned short* gptr, unsigned lds_off,
    int tile_k, int tile_n, int row_stride,
    int tensor_k, int tensor_n) {
  unsigned long long ga = (unsigned long long)gptr;
  u32x4 g0;
  g0[0] = 1u;                                            // count=1, user descriptor
  g0[1] = lds_off;                                       // LDS byte address
  g0[2] = (unsigned)(ga & 0xFFFFFFFFull);                // global_addr[31:0]
  g0[3] = (unsigned)((ga >> 32) & 0x1FFFFFFull) | (2u << 30);  // addr[56:32] | type=2
  i32x8 g1;
  g1[0] = (1 << 16);                                     // workgroup_mask=0, data_size=2B
  g1[1] = (int)(((unsigned)tensor_k & 0xFFFFu) << 16);   // tensor_dim0[15:0] @ bits 63:48
  g1[2] = (int)(((unsigned)tensor_k >> 16) |
                (((unsigned)tensor_n & 0xFFFFu) << 16)); // dim0 hi | dim1 lo
  g1[3] = (int)(((unsigned)tensor_n >> 16) |
                ((unsigned)tile_k << 16));               // dim1 hi | tile_dim0
  g1[4] = (int)(unsigned)tile_n;                         // tile_dim1, tile_dim2=0
  g1[5] = row_stride;                                    // tensor_dim0_stride[31:0]
  g1[6] = 0;                                             // stride0 hi | stride1 lo
  g1[7] = 0;                                             // stride1 hi
  i32x4 z4 = {};
#if __clang_major__ >= 23
  i32x8 z8 = {};
  __builtin_amdgcn_tensor_load_to_lds(g0, g1, z4, z4, z8, 0);
#else
  __builtin_amdgcn_tensor_load_to_lds(g0, g1, z4, z4, 0);
#endif
}

// ---------------------------------------------------------------- fp32 -> bf16
__global__ void cvt_f32_bf16(const float* __restrict__ src,
                             unsigned short* __restrict__ dst, int n) {
  int i = blockIdx.x * blockDim.x + threadIdx.x;
  int stride = gridDim.x * blockDim.x;
  for (; i < n; i += stride) dst[i] = f2bf(src[i]);
}

// ---------------------------------------------------------------- GEMM: C = A * W^T + bias
// A: [M][K] bf16 row-major, W: [N][K] bf16 row-major, C: [M][N] fp32.
// 256 threads = 8 waves; workgroup tile 128(M) x 64(N).
// B tile (64 x GEMM_BK) staged to LDS by the Tensor Data Mover, double-buffered,
// synchronized with TENSORcnt + workgroup barriers. A fragments register-direct.
// B fragments read back with explicit ds_load_b128 (TDM writes are invisible
// to the compiler's alias analysis, so reads must be opaque).
__global__ __launch_bounds__(256) void gemm_bf16_wmma(
    const unsigned short* __restrict__ A,
    const unsigned short* __restrict__ W,
    const float* __restrict__ bias,
    float* __restrict__ C,
    int M, int N, int K) {
  const int lane = threadIdx.x & 31;
  const int wave = threadIdx.x >> 5;
  const int hgrp = (lane >= 16);
  const int l16  = lane & 15;
  const int m0 = blockIdx.x * 128 + wave * 16;
  const int n0 = blockIdx.y * 64;

  // Only __shared__ object in this kernel -> LDS offset 0.
  __shared__ __align__(16) unsigned short Bt[2][64 * GEMM_BK];  // 2 x 16KB
  (void)Bt;
  const unsigned kBufBytes = 64 * GEMM_BK * 2;
  const int nsteps = K / GEMM_BK;

  const unsigned short* wtile = W + (size_t)n0 * K;
  if (wave == 0)                                    // prologue: fill buffer 0
    tdm_load_tile_bf16(wtile, 0, GEMM_BK, 64, K, K, N);

  v8f acc[4] = {};
  const unsigned short* arow = A + (size_t)(m0 + l16) * K;
  // per-lane byte offset of this lane's B row within a row block of the tile
  const unsigned laneRowB = (unsigned)l16 * GEMM_BK * 2 + (hgrp ? 32u : 0u);

  for (int step = 0; step < nsteps; ++step) {
    const int kbase = step * GEMM_BK;
    if (wave == 0) {
      if (step + 1 < nsteps) {                      // issue next, then wait current
        tdm_load_tile_bf16(wtile + kbase + GEMM_BK,
                           ((step + 1) & 1) * kBufBytes, GEMM_BK, 64, K, K, N);
        __builtin_amdgcn_s_wait_tensorcnt(1);       // current transfer landed
      } else {
        __builtin_amdgcn_s_wait_tensorcnt(0);       // drain on last step
      }
    }
    __syncthreads();                                // publish buf[step&1]
    const unsigned bufBase = (unsigned)(step & 1) * kBufBytes;
#pragma unroll
    for (int ks = 0; ks < GEMM_BK; ks += 32) {
      FragBF a;                                     // A: per-lane row, 2x16B chunks
      const unsigned short* ap = arow + kbase + ks + (hgrp ? 16 : 0);
      a.u4[0] = *(const u32x4*)(ap);
      a.u4[1] = *(const u32x4*)(ap + 8);
#pragma unroll
      for (int t = 0; t < 4; ++t) {                 // B: 32 contiguous bytes from LDS
        FragBF b;
        unsigned off = bufBase +
                       (unsigned)(t * 16) * GEMM_BK * 2 +  // row-block of tile
                       laneRowB +                          // lane's row + k-half
                       (unsigned)ks * 2;                   // k-substep
        lds_read_frag32(off, b.u4[0], b.u4[1]);
        acc[t] = __builtin_amdgcn_wmma_f32_16x16x32_bf16(
            false, a.v, false, b.v, (short)0, acc[t], false, false);
      }
    }
    __syncthreads();                                // done reading buf[step&1]
  }
#pragma unroll
  for (int t = 0; t < 4; ++t) {
    int col = n0 + t * 16 + l16;
    float bc = bias[col];
#pragma unroll
    for (int v = 0; v < 8; ++v) {
      int row = m0 + v + (hgrp ? 8 : 0);
      C[(size_t)row * N + col] = acc[t][v] + bc;
    }
  }
}

// ---------------------------------------------------------------- RMSNorm + RoPE + layout
// blockIdx.x = token, blockIdx.y = 0(q)/1(k)/2(v). 256 threads, 12 elems each.
__global__ __launch_bounds__(256) void norm_rope(
    const float* __restrict__ qf, const float* __restrict__ kf,
    const float* __restrict__ vf,
    const float* __restrict__ gq, const float* __restrict__ gk,
    const float* __restrict__ freqs, const int* __restrict__ grid_sizes,
    unsigned short* __restrict__ qh, unsigned short* __restrict__ kh,
    unsigned short* __restrict__ vt) {
  const int tok = blockIdx.x;
  const int which = blockIdx.y;
  const int tid = threadIdx.x;

  if (which == 2) {                       // V: bf16 + transpose -> [head][128][S]
    const float* row = vf + (size_t)tok * DIM;
#pragma unroll
    for (int j = 0; j < 12; ++j) {
      int e = tid * 12 + j;
      int h = e >> 7, d = e & 127;
      vt[(size_t)(h * HDIM + d) * SEQ + tok] = f2bf(row[e]);
    }
    return;
  }
  const float* row = (which == 0 ? qf : kf) + (size_t)tok * DIM;
  const float* g   = (which == 0) ? gq : gk;
  unsigned short* ob = (which == 0) ? qh : kh;

  float x[12];
  float ss = 0.f;
#pragma unroll
  for (int j = 0; j < 12; ++j) { x[j] = row[tid * 12 + j]; ss += x[j] * x[j]; }
  __shared__ float red[256];
  red[tid] = ss;
  __syncthreads();
  for (int off = 128; off > 0; off >>= 1) {
    if (tid < off) red[tid] += red[tid + off];
    __syncthreads();
  }
  float inv = rsqrtf(red[0] / (float)DIM + EPSF);

  int gh = grid_sizes[1], gw = grid_sizes[2];
  int fi = tok / (gh * gw);
  int rem = tok % (gh * gw);
  int hi = rem / gw, wi = rem % gw;

#pragma unroll
  for (int p = 0; p < 6; ++p) {
    int e0 = tid * 12 + 2 * p;
    int h = e0 >> 7, d = e0 & 127;
    int c = d >> 1;
    int pos = (c < CF_P) ? fi : ((c < CF_P + CH_P) ? hi : wi);
    float th = freqs[pos * C2TAB + c];
    float sn, cs;
    __sincosf(th, &sn, &cs);
    float xr = x[2 * p]     * inv * g[e0];
    float xi = x[2 * p + 1] * inv * g[e0 + 1];
    size_t base = ((size_t)h * SEQ + tok) * HDIM + d;
    ob[base]     = f2bf(xr * cs - xi * sn);
    ob[base + 1] = f2bf(xr * sn + xi * cs);
  }
}

// ---------------------------------------------------------------- flash attention
// grid = (SEQ/64, NHEADS), block = 128 (4 waves); wave owns 16 query rows.
__global__ __launch_bounds__(128) void attn_wmma(
    const unsigned short* __restrict__ qh,   // [H][S][128]
    const unsigned short* __restrict__ kh,   // [H][S][128]
    const unsigned short* __restrict__ vt,   // [H][128][S]
    const int* __restrict__ seq_lens,
    unsigned short* __restrict__ attnb) {    // [S][DIM] bf16
  const int lane = threadIdx.x & 31;
  const int wave = threadIdx.x >> 5;
  const int hgrp = (lane >= 16);
  const int l16  = lane & 15;
  const int head = blockIdx.y;
  const int q0 = blockIdx.x * 64 + wave * 16;
  const int kvlen = seq_lens[0];
  const float scale = 0.08838834764831845f;  // 1/sqrt(128)

  const unsigned short* qbase = qh + (size_t)head * SEQ * HDIM;
  const unsigned short* kbase = kh + (size_t)head * SEQ * HDIM;
  const unsigned short* vbase = vt + (size_t)head * HDIM * SEQ;

  FragBF aq[4];                         // Q row fragment, 4 d-chunks of 32
#pragma unroll
  for (int j = 0; j < 4; ++j) {
    const unsigned short* p =
        qbase + (size_t)(q0 + l16) * HDIM + j * 32 + (hgrp ? 16 : 0);
    aq[j].u4[0] = *(const u32x4*)p;
    aq[j].u4[1] = *(const u32x4*)(p + 8);
  }

  v8f acc[8] = {};
  float mstat[8], lstat[8];
#pragma unroll
  for (int v = 0; v < 8; ++v) { mstat[v] = -1e30f; lstat[v] = 0.f; }

  __shared__ __align__(16) unsigned short Plds[4][16 * 32];
  unsigned short* pw = &Plds[wave][0];

  for (int k0 = 0; k0 < SEQ; k0 += 32) {
    v8f s0 = {}, s1 = {};
#pragma unroll
    for (int j = 0; j < 4; ++j) {       // S = Q * K^T (two 16-col halves)
      FragBF b0, b1;
      const unsigned short* kp0 =
          kbase + (size_t)(k0 + l16) * HDIM + j * 32 + (hgrp ? 16 : 0);
      const unsigned short* kp1 =
          kbase + (size_t)(k0 + 16 + l16) * HDIM + j * 32 + (hgrp ? 16 : 0);
      b0.u4[0] = *(const u32x4*)kp0; b0.u4[1] = *(const u32x4*)(kp0 + 8);
      b1.u4[0] = *(const u32x4*)kp1; b1.u4[1] = *(const u32x4*)(kp1 + 8);
      s0 = __builtin_amdgcn_wmma_f32_16x16x32_bf16(false, aq[j].v, false, b0.v,
                                                   (short)0, s0, false, false);
      s1 = __builtin_amdgcn_wmma_f32_16x16x32_bf16(false, aq[j].v, false, b1.v,
                                                   (short)0, s1, false, false);
    }
    bool ok0 = (k0 + l16) < kvlen;
    bool ok1 = (k0 + 16 + l16) < kvlen;
    float pr0[8], pr1[8], alpha[8];
#pragma unroll
    for (int v = 0; v < 8; ++v) {       // online softmax, per row of the tile
      float a0 = ok0 ? s0[v] * scale : -1e9f;
      float a1 = ok1 ? s1[v] * scale : -1e9f;
      float mx = fmaxf(a0, a1);
      mx = fmaxf(mx, __shfl_xor(mx, 1, 32));
      mx = fmaxf(mx, __shfl_xor(mx, 2, 32));
      mx = fmaxf(mx, __shfl_xor(mx, 4, 32));
      mx = fmaxf(mx, __shfl_xor(mx, 8, 32));
      float mnew = fmaxf(mstat[v], mx);
      alpha[v] = __expf(mstat[v] - mnew);
      pr0[v] = __expf(a0 - mnew);
      pr1[v] = __expf(a1 - mnew);
      float rs = pr0[v] + pr1[v];
      rs += __shfl_xor(rs, 1, 32);
      rs += __shfl_xor(rs, 2, 32);
      rs += __shfl_xor(rs, 4, 32);
      rs += __shfl_xor(rs, 8, 32);
      lstat[v] = lstat[v] * alpha[v] + rs;
      mstat[v] = mnew;
    }
#pragma unroll
    for (int t = 0; t < 8; ++t)
#pragma unroll
      for (int v = 0; v < 8; ++v) acc[t][v] *= alpha[v];

    // C-layout -> A-layout via per-wave LDS staging of P (bf16 16x32 tile)
#pragma unroll
    for (int v = 0; v < 8; ++v) {
      int m = v + (hgrp ? 8 : 0);
      pw[m * 32 + l16]      = f2bf(pr0[v]);
      pw[m * 32 + 16 + l16] = f2bf(pr1[v]);
    }
    asm volatile("s_wait_dscnt 0" ::: "memory");  // in-wave LDS RAW (CDNA5 split counter)
    FragBF pa;
    const unsigned short* prd = pw + l16 * 32 + (hgrp ? 16 : 0);
    pa.u4[0] = *(const u32x4*)prd;
    pa.u4[1] = *(const u32x4*)(prd + 8);

#pragma unroll
    for (int t = 0; t < 8; ++t) {       // O += P * V (V transposed -> contiguous B)
      FragBF bvf;
      const unsigned short* vp =
          vbase + (size_t)(t * 16 + l16) * SEQ + k0 + (hgrp ? 16 : 0);
      bvf.u4[0] = *(const u32x4*)vp;
      bvf.u4[1] = *(const u32x4*)(vp + 8);
      acc[t] = __builtin_amdgcn_wmma_f32_16x16x32_bf16(
          false, pa.v, false, bvf.v, (short)0, acc[t], false, false);
    }
  }
#pragma unroll
  for (int t = 0; t < 8; ++t) {
    int d = t * 16 + l16;
#pragma unroll
    for (int v = 0; v < 8; ++v) {
      int row = q0 + v + (hgrp ? 8 : 0);
      attnb[(size_t)row * DIM + head * HDIM + d] = f2bf(acc[t][v] / lstat[v]);
    }
  }
}

// ---------------------------------------------------------------- launch
extern "C" void kernel_launch(void* const* d_in, const int* in_sizes, int n_in,
                              void* d_out, int out_size, void* d_ws, size_t ws_size,
                              hipStream_t stream) {
  const float* x          = (const float*)d_in[0];
  const int*   seq_lens   = (const int*)d_in[1];
  const int*   grid_sizes = (const int*)d_in[2];
  const float* freqs      = (const float*)d_in[3];
  const float* wq = (const float*)d_in[4];
  const float* bq = (const float*)d_in[5];
  const float* wk = (const float*)d_in[6];
  const float* bk = (const float*)d_in[7];
  const float* wv = (const float*)d_in[8];
  const float* bv = (const float*)d_in[9];
  const float* wo = (const float*)d_in[10];
  const float* bo = (const float*)d_in[11];
  const float* gq = (const float*)d_in[12];
  const float* gk = (const float*)d_in[13];
  float* out = (float*)d_out;

  const size_t SD = (size_t)SEQ * DIM;
  const size_t DD = (size_t)DIM * DIM;
  char* ws = (char*)d_ws;
  size_t off = 0;
  auto alloc = [&](size_t bytes) -> void* {
    void* p = ws + off;
    off = (off + bytes + 255) & ~(size_t)255;
    return p;
  };
  unsigned short* xb    = (unsigned short*)alloc(SD * 2);
  unsigned short* wqb   = (unsigned short*)alloc(DD * 2);
  unsigned short* wkb   = (unsigned short*)alloc(DD * 2);
  unsigned short* wvb   = (unsigned short*)alloc(DD * 2);
  unsigned short* wob   = (unsigned short*)alloc(DD * 2);
  float*          qf    = (float*)alloc(SD * 4);
  float*          kf    = (float*)alloc(SD * 4);
  float*          vf    = (float*)alloc(SD * 4);
  unsigned short* qhb   = (unsigned short*)alloc(SD * 2);
  unsigned short* khb   = (unsigned short*)alloc(SD * 2);
  unsigned short* vtb   = (unsigned short*)alloc(SD * 2);
  unsigned short* attnb = (unsigned short*)alloc(SD * 2);

  cvt_f32_bf16<<<dim3(1024), 256, 0, stream>>>(x,  xb,  (int)SD);
  cvt_f32_bf16<<<dim3(2048), 256, 0, stream>>>(wq, wqb, (int)DD);
  cvt_f32_bf16<<<dim3(2048), 256, 0, stream>>>(wk, wkb, (int)DD);
  cvt_f32_bf16<<<dim3(2048), 256, 0, stream>>>(wv, wvb, (int)DD);
  cvt_f32_bf16<<<dim3(2048), 256, 0, stream>>>(wo, wob, (int)DD);

  dim3 ggrid(SEQ / 128, DIM / 64);
  gemm_bf16_wmma<<<ggrid, 256, 0, stream>>>(xb, wqb, bq, qf, SEQ, DIM, DIM);
  gemm_bf16_wmma<<<ggrid, 256, 0, stream>>>(xb, wkb, bk, kf, SEQ, DIM, DIM);
  gemm_bf16_wmma<<<ggrid, 256, 0, stream>>>(xb, wvb, bv, vf, SEQ, DIM, DIM);

  norm_rope<<<dim3(SEQ, 3), 256, 0, stream>>>(qf, kf, vf, gq, gk, freqs,
                                              grid_sizes, qhb, khb, vtb);

  attn_wmma<<<dim3(SEQ / 64, NHEADS), 128, 0, stream>>>(qhb, khb, vtb,
                                                        seq_lens, attnb);

  gemm_bf16_wmma<<<ggrid, 256, 0, stream>>>(attnb, wob, bo, out, SEQ, DIM, DIM);
}
